// GNNEncoder_70153995812874
// MI455X (gfx1250) — matrix operand order repaired and checked by reading, not compiled, not run on hardware
//
#include <hip/hip_runtime.h>
#include <hip/hip_bf16.h>

// ---------------------------------------------------------------------------
// GNN encoder for MI455X (gfx1250, wave32, WMMA).
// Edge MLP = two bf16 WMMA GEMMs per 16-edge tile; scatter via f32 atomics.
// Whole working set (10MB bf16 features + 5MB edge index) is L2-resident, so
// the edge MLP is compute-bound -> v_wmma_f32_16x16x32_bf16 is the right tool.
// ---------------------------------------------------------------------------

typedef __attribute__((ext_vector_type(16))) __bf16 v16bf;
typedef __attribute__((ext_vector_type(8)))  float  v8f;

union FragBF { v16bf v; uint4 q[2]; };

#define FEAT       128            // F_IN == HID == 128
#define KIN        256            // concat(dst,src) = 2*128
#define NGRAPH     64
#define LATOUT     512            // LAT * N_SUPER

// LDS layout for the edge kernel (rows padded +8 bf16 to rotate banks by 4r)
#define W1_ROW     (KIN + 8)      // 264 bf16 per row
#define W2_ROW     (FEAT + 8)     // 136 bf16 per row
#define ST_ROW     (FEAT + 8)     // 136 bf16 per staged row
#define SM_W1_OFF  0
#define SM_W2_OFF  (SM_W1_OFF + 128 * W1_ROW * 2)           // 67584
#define SM_B1_OFF  (SM_W2_OFF + 128 * W2_ROW * 2)           // 102400
#define SM_B2_OFF  (SM_B1_OFF + 512)                        // 102912
#define SM_ST_OFF  (SM_B2_OFF + 512)                        // 103424
#define SM_BYTES   (SM_ST_OFF + 8 * 16 * ST_ROW * 2)        // 138240

// ---------------------------- utility kernels ------------------------------

__global__ void k_zero_f32(float* __restrict__ p, int n) {
  int i = blockIdx.x * blockDim.x + threadIdx.x;
  if (i < n) p[i] = 0.0f;
}

__global__ void k_f32_to_bf16(const float* __restrict__ in, __bf16* __restrict__ out, int n) {
  int i = blockIdx.x * blockDim.x + threadIdx.x;
  if (i < n) out[i] = (__bf16)in[i];
}

__global__ void k_relu_f32_to_bf16(const float* __restrict__ in, __bf16* __restrict__ out, int n) {
  int i = blockIdx.x * blockDim.x + threadIdx.x;
  if (i < n) { float v = in[i]; out[i] = (__bf16)(v > 0.0f ? v : 0.0f); }
}

// w: [K][N] row-major f32  ->  wt: [N][K] row-major bf16 (B == W^T so B-frag
// loads use the same per-lane row pattern as A-frag loads)
__global__ void k_transpose_to_bf16(const float* __restrict__ w, __bf16* __restrict__ wt,
                                    int K, int N) {
  int i = blockIdx.x * blockDim.x + threadIdx.x;
  if (i < K * N) {
    int k = i / N, n = i - k * N;
    wt[n * K + k] = (__bf16)w[i];
  }
}

// ----------------------------- edge MLP kernel -----------------------------
// Per wave: one tile of 16 edges.
//   A  (16 x 256 bf16)  = gather of feat[dst]||feat[src]   (direct from global/L2)
//   Z1 (16 x 128 f32)   = relu(A @ W1^T' + b1)   (8 k-tiles x 8 n-tiles WMMA)
//   Z2 (16 x 128 f32)   = Z1bf @ W2^T' + b2      (4 k-tiles x 8 n-tiles WMMA)
//   scatter-add Z2 rows into out[dst[e]] with global_atomic_add_f32
//
// Fragment layouts per CDNA5 ISA 7.12.2 (16-bit, wave32):
//   A 16x32 : lane = m + 16*hi ; elems 0..7 -> K = hi*8..hi*8+7,
//             elems 8..15 -> K = 16+hi*8 .. 23+hi*8
//   B 32x16 : lane = n + 16*hi ; elems 0..7 -> K = hi*8..hi*8+7, 8..15 -> K = 16+hi*8 .. 23+hi*8
//   C/D 16x16 f32: lane = n + 16*hi ; vgpr v -> m = v + 8*hi
__global__ void __launch_bounds__(256)
k_edge_mlp(const __bf16* __restrict__ feat,     // [N,128] bf16
           const int*    __restrict__ src,      // [E]
           const int*    __restrict__ dst,      // [E]
           const __bf16* __restrict__ wt1,      // [128][256] bf16  (= W1^T)
           const float*  __restrict__ b1,       // [128]
           const __bf16* __restrict__ wt2,      // [128][128] bf16  (= W2^T)
           const float*  __restrict__ b2,       // [128]
           float*        __restrict__ out,      // [N,128] f32 accumulator (pre-zeroed)
           int nTiles)
{
  extern __shared__ char smem[];
  __bf16* sW1 = (__bf16*)(smem + SM_W1_OFF);
  __bf16* sW2 = (__bf16*)(smem + SM_W2_OFF);
  float*  sB1 = (float*) (smem + SM_B1_OFF);
  float*  sB2 = (float*) (smem + SM_B2_OFF);
  __bf16* sStage = (__bf16*)(smem + SM_ST_OFF);

  // cooperative weight load into padded LDS (16B chunks)
  for (int i = threadIdx.x; i < 128 * (KIN / 8); i += 256) {   // 128 rows x 32 uint4
    int row = i >> 5, off = i & 31;
    *(uint4*)(sW1 + row * W1_ROW + off * 8) = ((const uint4*)wt1)[row * 32 + off];
  }
  for (int i = threadIdx.x; i < 128 * (FEAT / 8); i += 256) {  // 128 rows x 16 uint4
    int row = i >> 4, off = i & 15;
    *(uint4*)(sW2 + row * W2_ROW + off * 8) = ((const uint4*)wt2)[row * 16 + off];
  }
  for (int i = threadIdx.x; i < 128; i += 256) { sB1[i] = b1[i]; sB2[i] = b2[i]; }
  __syncthreads();

  const int wave  = threadIdx.x >> 5;
  const int lane  = threadIdx.x & 31;
  const int r     = lane & 15;          // row (A) / col (B,C) within tile
  const int hi    = lane >> 4;          // lane-half selector
  const int gwave = blockIdx.x * 8 + wave;
  const int nWav  = gridDim.x * 8;
  __bf16* st = sStage + wave * (16 * ST_ROW);

  const v8f vzero = {0.f, 0.f, 0.f, 0.f, 0.f, 0.f, 0.f, 0.f};

  for (int tile = gwave; tile < nTiles; tile += nWav) {
    const int e0 = tile * 16;
    const int myDst = dst[e0 + r];      // lanes 0..15 hold rows 0..15 (dup in 16..31)
    const int mySrc = src[e0 + r];

    // ---------------- GEMM1: [16x256] x [256x128] ----------------
    v8f acc[8];
#pragma unroll
    for (int nt = 0; nt < 8; ++nt) acc[nt] = vzero;

#pragma unroll
    for (int kt = 0; kt < 8; ++kt) {
      const int k0   = kt * 32;
      const int node = (k0 < FEAT) ? myDst : mySrc;   // concat(dst||src)
      const int kloc = k0 & (FEAT - 1);
      const __bf16* ap = feat + (size_t)node * FEAT + kloc + hi * 8;
      FragBF a;
      a.q[0] = *(const uint4*)(ap);        // K = kh .. kh+7
      a.q[1] = *(const uint4*)(ap + 16);   // K = kh+16 .. kh+23
#pragma unroll
      for (int nt = 0; nt < 8; ++nt) {
        const __bf16* bp = sW1 + (nt * 16 + r) * W1_ROW + k0 + hi * 8;
        FragBF b;
        b.q[0] = *(const uint4*)(bp);
        b.q[1] = *(const uint4*)(bp + 16);
        acc[nt] = __builtin_amdgcn_wmma_f32_16x16x32_bf16(
            false, a.v, false, b.v, (short)0, acc[nt], false, false);
      }
    }

    // bias + relu, restage as bf16 rows for the second GEMM
#pragma unroll
    for (int nt = 0; nt < 8; ++nt) {
      const int col = nt * 16 + r;
      const float bias = sB1[col];
#pragma unroll
      for (int v = 0; v < 8; ++v) {
        const int m = v + hi * 8;
        float z = acc[nt][v] + bias;
        z = z > 0.0f ? z : 0.0f;
        st[m * ST_ROW + col] = (__bf16)z;
      }
    }
    // same-wave LDS RAW: DS ops are in-order, but force the wait + stop
    // compiler reordering before we read the staged tile back.
    asm volatile("s_wait_dscnt 0" ::: "memory");

    // ---------------- GEMM2: [16x128] x [128x128] ----------------
    v8f acc2[8];
#pragma unroll
    for (int nt = 0; nt < 8; ++nt) acc2[nt] = vzero;

#pragma unroll
    for (int kt = 0; kt < 4; ++kt) {
      const int k0 = kt * 32;
      const __bf16* ap = st + r * ST_ROW + k0 + hi * 8;
      FragBF a;
      a.q[0] = *(const uint4*)(ap);
      a.q[1] = *(const uint4*)(ap + 16);
#pragma unroll
      for (int nt = 0; nt < 8; ++nt) {
        const __bf16* bp = sW2 + (nt * 16 + r) * W2_ROW + k0 + hi * 8;
        FragBF b;
        b.q[0] = *(const uint4*)(bp);
        b.q[1] = *(const uint4*)(bp + 16);
        acc2[nt] = __builtin_amdgcn_wmma_f32_16x16x32_bf16(
            false, a.v, false, b.v, (short)0, acc2[nt], false, false);
      }
    }
    asm volatile("" ::: "memory");   // keep stage reads before next-iter stores

    // scatter-add msg rows (+b2) into out[dst]
#pragma unroll
    for (int nt = 0; nt < 8; ++nt) {
      const int col = nt * 16 + r;
      const float bias = sB2[col];
#pragma unroll
      for (int v = 0; v < 8; ++v) {
        const int m = v + hi * 8;
        const int node = __shfl(myDst, m, 32);
        atomicAdd(out + (size_t)node * FEAT + col, acc2[nt][v] + bias);
      }
    }
  }
}

// ------------------------------ pooling ------------------------------------

// sums relu(h) per graph + counts (batch is int32 per harness convention)
__global__ void k_pool(const float* __restrict__ h, const int* __restrict__ batch,
                       float* __restrict__ pooled, float* __restrict__ cnt, int nNodes) {
  int i = blockIdx.x * blockDim.x + threadIdx.x;     // node*32 + q (4 feats each)
  if (i >= nNodes * 32) return;
  int node = i >> 5, q = i & 31;
  int g = batch[node];
  const float4 v = *(const float4*)(h + (size_t)node * FEAT + q * 4);
  float* base = pooled + g * FEAT + q * 4;
  atomicAdd(base + 0, v.x > 0.f ? v.x : 0.f);
  atomicAdd(base + 1, v.y > 0.f ? v.y : 0.f);
  atomicAdd(base + 2, v.z > 0.f ? v.z : 0.f);
  atomicAdd(base + 3, v.w > 0.f ? v.w : 0.f);
  if (q == 0) atomicAdd(cnt + g, 1.0f);
}

__global__ void k_mean(float* __restrict__ pooled, const float* __restrict__ cnt, int n) {
  int i = blockIdx.x * blockDim.x + threadIdx.x;
  if (i < n) {
    float c = cnt[i >> 7];
    pooled[i] /= (c > 1.0f ? c : 1.0f);
  }
}

// head MLP: latent = relu(pooled@wm1+bm1)@wm2+bm2   (tiny; fp32 scalar)
__global__ void __launch_bounds__(256)
k_head(const float* __restrict__ pooled,   // [64][128] (already mean)
       const float* __restrict__ wm1, const float* __restrict__ bm1,
       const float* __restrict__ wm2, const float* __restrict__ bm2,
       float* __restrict__ out) {          // [64][512]
  __shared__ float sH[NGRAPH * FEAT];      // 32 KB
  const int tid = threadIdx.x;
  for (int i = tid; i < NGRAPH * FEAT; i += 256) {
    int g = i >> 7, n = i & 127;
    float acc = bm1[n];
    const float* p = pooled + g * FEAT;
#pragma unroll 4
    for (int k = 0; k < FEAT; ++k) acc += p[k] * wm1[k * FEAT + n];
    sH[i] = acc > 0.0f ? acc : 0.0f;
  }
  __syncthreads();
  for (int i = tid; i < NGRAPH * LATOUT; i += 256) {
    int g = i >> 9, n = i & 511;
    float acc = bm2[n];
    const float* hrow = sH + g * FEAT;
#pragma unroll 4
    for (int k = 0; k < FEAT; ++k) acc += hrow[k] * wm2[k * LATOUT + n];
    out[i] = acc;
  }
}

// ------------------------------ launcher -----------------------------------

extern "C" void kernel_launch(void* const* d_in, const int* in_sizes, int n_in,
                              void* d_out, int out_size, void* d_ws, size_t ws_size,
                              hipStream_t stream) {
  const float* x    = (const float*)d_in[0];
  const int*   eidx = (const int*)  d_in[1];
  const int*   batch= (const int*)  d_in[2];
  const float* w11  = (const float*)d_in[3];
  const float* b11  = (const float*)d_in[4];
  const float* w12  = (const float*)d_in[5];
  const float* b12  = (const float*)d_in[6];
  const float* w21  = (const float*)d_in[7];
  const float* b21  = (const float*)d_in[8];
  const float* w22  = (const float*)d_in[9];
  const float* b22  = (const float*)d_in[10];
  const float* wm1  = (const float*)d_in[11];
  const float* bm1  = (const float*)d_in[12];
  const float* wm2  = (const float*)d_in[13];
  const float* bm2  = (const float*)d_in[14];

  const int N = in_sizes[0] / FEAT;     // 40000
  const int E = in_sizes[1] / 2;        // 640000
  const int* src = eidx;                // edge_index[0]
  const int* dst = eidx + E;            // edge_index[1]
  const int nTiles = E / 16;            // E is a multiple of 16

  // workspace carve-up (256B aligned)
  char* ws = (char*)d_ws;
  size_t off = 0;
  auto carve = [&](size_t bytes) -> char* {
    char* p = ws + off;
    off += (bytes + 255) & ~(size_t)255;
    return p;
  };
  __bf16* xh    = (__bf16*)carve((size_t)N * FEAT * 2);
  __bf16* h1h   = (__bf16*)carve((size_t)N * FEAT * 2);
  float*  h1f   = (float*) carve((size_t)N * FEAT * 4);
  float*  h2f   = (float*) carve((size_t)N * FEAT * 4);
  __bf16* wt1a  = (__bf16*)carve(128 * KIN  * 2);
  __bf16* wt1b  = (__bf16*)carve(128 * FEAT * 2);
  __bf16* wt2a  = (__bf16*)carve(128 * KIN  * 2);
  __bf16* wt2b  = (__bf16*)carve(128 * FEAT * 2);
  float*  pooled= (float*) carve(NGRAPH * FEAT * 4 + NGRAPH * 4); // pooled + cnt contig
  float*  cnt   = pooled + NGRAPH * FEAT;
  (void)ws_size; (void)n_in; (void)out_size;

  // allow >64KB dynamic LDS for the edge kernel (host-side, capture-safe)
  (void)hipFuncSetAttribute((const void*)k_edge_mlp,
                            hipFuncAttributeMaxDynamicSharedMemorySize, SM_BYTES);

  const int T = 256;
  const int nFeatEl = N * FEAT;

  // 1) precision conversion: features + transposed weights -> bf16
  k_f32_to_bf16<<<(nFeatEl + T - 1) / T, T, 0, stream>>>(x, xh, nFeatEl);
  k_transpose_to_bf16<<<(KIN  * FEAT + T - 1) / T, T, 0, stream>>>(w11, wt1a, KIN,  FEAT);
  k_transpose_to_bf16<<<(FEAT * FEAT + T - 1) / T, T, 0, stream>>>(w12, wt1b, FEAT, FEAT);
  k_transpose_to_bf16<<<(KIN  * FEAT + T - 1) / T, T, 0, stream>>>(w21, wt2a, KIN,  FEAT);
  k_transpose_to_bf16<<<(FEAT * FEAT + T - 1) / T, T, 0, stream>>>(w22, wt2b, FEAT, FEAT);

  // 2) layer 1
  k_zero_f32<<<(nFeatEl + T - 1) / T, T, 0, stream>>>(h1f, nFeatEl);
  k_edge_mlp<<<512, 256, SM_BYTES, stream>>>(xh, src, dst, wt1a, b11, wt1b, b12, h1f, nTiles);
  k_relu_f32_to_bf16<<<(nFeatEl + T - 1) / T, T, 0, stream>>>(h1f, h1h, nFeatEl);

  // 3) layer 2
  k_zero_f32<<<(nFeatEl + T - 1) / T, T, 0, stream>>>(h2f, nFeatEl);
  k_edge_mlp<<<512, 256, SM_BYTES, stream>>>(h1h, src, dst, wt2a, b21, wt2b, b22, h2f, nTiles);

  // 4) mean pool (relu fused into pool reads) + head MLP
  k_zero_f32<<<(NGRAPH * FEAT + NGRAPH + T - 1) / T, T, 0, stream>>>(pooled, NGRAPH * FEAT + NGRAPH);
  k_pool<<<(N * 32 + T - 1) / T, T, 0, stream>>>(h2f, batch, pooled, cnt, N);
  k_mean<<<(NGRAPH * FEAT + T - 1) / T, T, 0, stream>>>(pooled, cnt, NGRAPH * FEAT);
  k_head<<<1, 256, 0, stream>>>(pooled, wm1, bm1, wm2, bm2, (float*)d_out);
}
